// InformationDiffusion_33775622815870
// MI455X (gfx1250) — matrix operand order repaired and checked by reading, not compiled, not run on hardware
//
#include <hip/hip_runtime.h>
#include <cstdint>
#include <cstddef>

// ---------------------------------------------------------------------------
// Types / WMMA helpers (CDNA5 gfx1250, wave32)
// ---------------------------------------------------------------------------
typedef __attribute__((ext_vector_type(16))) _Float16     v16h;
typedef __attribute__((ext_vector_type(8)))  float        v8f;
typedef __attribute__((ext_vector_type(4)))  unsigned int v4u;
typedef int v4i __attribute__((__vector_size__(16)));   // <4 x i32> for async builtin

#define DEVI __device__ __forceinline__

union HU { v16h h; v4u q[2]; };

DEVI v8f do_wmma(v16h a, v16h b, v8f c) {
  // D = A(16x32 f16) * B(32x16 f16) + C(16x16 f32)
  return __builtin_amdgcn_wmma_f32_16x16x32_f16(false, a, false, b, (short)0, c,
                                                false, false);
}

DEVI v16h zfrag() { v16h z = {}; return z; }

// ---------------------------------------------------------------------------
// Async global->LDS staging (CDNA5 GLOBAL_LOAD_ASYNC_TO_LDS_B128, ASYNCcnt)
// with sync fallback if the builtin is unavailable on this toolchain.
// ---------------------------------------------------------------------------
#if defined(__HIP_DEVICE_COMPILE__) && \
    __has_builtin(__builtin_amdgcn_global_load_async_to_lds_b128)
#define ASYNC_STAGE 1
#else
#define ASYNC_STAGE 0
#endif

DEVI void stage16B(const _Float16* g, _Float16* l) {
#if ASYNC_STAGE
  __builtin_amdgcn_global_load_async_to_lds_b128(
      (__attribute__((address_space(1))) v4i*)g,
      (__attribute__((address_space(3))) v4i*)l, 0, 0);
#else
  *(v4u*)l = *(const v4u*)g;
#endif
}

DEVI void wait_stage() {
#if ASYNC_STAGE
#if __has_builtin(__builtin_amdgcn_s_wait_asynccnt)
  __builtin_amdgcn_s_wait_asynccnt(0);
#else
  asm volatile("s_wait_asynccnt 0" ::: "memory");
#endif
#endif
}

// A fragment: 16x32 f16. lane&15 = row, lane>>4 selects K-halves:
// elements 0..7 = K[h*8 .. h*8+7], elements 8..15 = K[16+h*8 .. 16+h*8+7]
DEVI v16h ldA(const _Float16* A, int ldk, int mbase, int kb) {
  int lane = threadIdx.x & 31;
  const _Float16* p = A + (size_t)(mbase + (lane & 15)) * ldk + kb + ((lane >> 4) << 3);
  HU u; u.q[0] = *(const v4u*)p; u.q[1] = *(const v4u*)(p + 16);
  return u.h;
}

// B fragment: 32x16 f16, source stored [N][K] (K contiguous).
// lane&15 = column n, elements 0..7 = K[h*8..h*8+7], 8..15 = K[16+h*8..16+h*8+7]
DEVI v16h ldB(const _Float16* B, int ldk, int nbase, int kb) {
  int lane = threadIdx.x & 31;
  const _Float16* p = B + (size_t)(nbase + (lane & 15)) * ldk + kb + ((lane >> 4) << 3);
  HU u; u.q[0] = *(const v4u*)p; u.q[1] = *(const v4u*)(p + 16);
  return u.h;
}

// B fragment from LDS stage Bs[64][32] (f16)
DEVI v16h ldBs(const _Float16* Bs, int nt) {
  int lane = threadIdx.x & 31;
  const _Float16* p = Bs + ((nt << 4) + (lane & 15)) * 32 + ((lane >> 4) << 3);
  HU u; u.q[0] = *(const v4u*)p; u.q[1] = *(const v4u*)(p + 16);
  return u.h;
}

// ---------------------------------------------------------------------------
// Problem constants
// ---------------------------------------------------------------------------
#define CF   256
#define HW   4096
#define WIM  64
#define HID  64
#define NSTEP 4

// ---------------------------------------------------------------------------
// Workspace layout (bytes, all 256-aligned)
// ---------------------------------------------------------------------------
static constexpr size_t OFF_WC1  = 0;                         // 64x512 f16
static constexpr size_t OFF_WC2  = OFF_WC1  + 65536;          // 256x64 f16
static constexpr size_t OFF_WNL1 = OFF_WC2  + 32768;          // 4x9x256x256 f16
static constexpr size_t OFF_WNL2 = OFF_WNL1 + 4718592;
static constexpr size_t OFF_WQKV = OFF_WNL2 + 4718592;        // 768x256 f16
static constexpr size_t OFF_WWO  = OFF_WQKV + 393216;         // 256x256 f16
static constexpr size_t OFF_WFP1 = OFF_WWO  + 131072;         // 9x256x256 f16
static constexpr size_t OFF_WFP2 = OFF_WFP1 + 1179648;        // 256x256 f16
static constexpr size_t OFF_GRID = OFF_WFP2 + 131072;         // [C][HW] f32
static constexpr size_t OFF_ORIG = OFF_GRID + (4u << 20);
static constexpr size_t OFF_MASK = OFF_ORIG + (4u << 20);     // [HW] f32
static constexpr size_t OFF_GST  = OFF_MASK + 16384;          // 16 f32
static constexpr size_t OFF_SCR  = OFF_GST  + 256;
// diffusion-stage scratch (within SCR)
static constexpr size_t OFF_GM   = OFF_SCR + 0;               // f32 [C][HW]  (alias r1)
static constexpr size_t OFF_LAP  = OFF_SCR + (4u  << 20);     // f32 [C][HW]  (alias r2)
static constexpr size_t OFF_CX   = OFF_SCR + (8u  << 20);     // f16 [HW][512] / f32 cond alias
static constexpr size_t OFF_H16  = OFF_SCR + (12u << 20);     // f16 [HW][64]
static constexpr size_t OFF_X    = OFF_SCR + (13u << 20);     // f32 [C][HW]
static constexpr size_t OFF_XT   = OFF_SCR + (17u << 20);     // f16 [HW][C]
static constexpr size_t OFF_RT   = OFF_SCR + (19u << 20);     // f16 [HW][C]
// attention/final-stage scratch (re-uses SCR)
static constexpr size_t OFF_SEQT = OFF_SCR + 0;               // f16 [HW][C]
static constexpr size_t OFF_QKV  = OFF_SCR + (2u  << 20);     // f16 [HW][768]
static constexpr size_t OFF_VT   = OFF_SCR + (8u  << 20);     // f16 [8][64][HW]
static constexpr size_t OFF_OT   = OFF_SCR + (12u << 20);     // f16 [HW][C]
static constexpr size_t OFF_O2   = OFF_SCR + (14u << 20);     // f32 [HW][C]
static constexpr size_t OFF_LNT  = OFF_SCR + 0;               // f16 [HW][C]   (seqt dead)
static constexpr size_t OFF_Y    = OFF_SCR + (2u  << 20);     // f32 [C][HW]   (qkv dead)
static constexpr size_t OFF_YT   = OFF_SCR + (6u  << 20);     // f16 [HW][C]

// ---------------------------------------------------------------------------
// Elementwise / prep kernels
// ---------------------------------------------------------------------------
__global__ void k_zero(float* p, int n) {
  int i = blockIdx.x * 256 + threadIdx.x;
  if (i < n) p[i] = 0.f;
}

__global__ void k_cast16(const float* __restrict__ s, _Float16* __restrict__ d, int n) {
  int i = blockIdx.x * 256 + threadIdx.x;
  if (i < n) d[i] = (_Float16)s[i];
}

// src (S,O=256,I=256,3,3) -> dst [s][tap][o][i] f16
__global__ void k_repack_conv(const float* __restrict__ src, _Float16* __restrict__ dst,
                              int total) {
  int tid = blockIdx.x * 256 + threadIdx.x;
  if (tid >= total) return;
  int i   = tid & 255;
  int o   = (tid >> 8) & 255;
  int rem = tid >> 16;           // s*9 + tap
  int tap = rem % 9;
  int s   = rem / 9;
  dst[tid] = (_Float16)src[((((size_t)s * 256 + o) * 256 + i) * 9) + tap];
}

__global__ void k_scatter(const float* __restrict__ sf, const int* __restrict__ pos,
                          float* __restrict__ grid, float* __restrict__ orig,
                          float* __restrict__ mask, int n) {
  int tid = blockIdx.x * 256 + threadIdx.x;
  if (tid >= n) return;
  int nn = tid >> 8, c = tid & 255;
  int p = pos[nn * 2] * WIM + pos[nn * 2 + 1];
  float v = sf[(size_t)nn * 256 + c];
  grid[(size_t)c * HW + p] = v;
  orig[(size_t)c * HW + p] = v;
  if (c == 0) mask[p] = 1.f;
}

// Sobel-x/y magnitude + Laplacian + pack [gm;grid] transposed f16
__global__ void k_stencil(const float* __restrict__ grid, float* __restrict__ gm,
                          float* __restrict__ lap, _Float16* __restrict__ condX) {
  int tid = blockIdx.x * 256 + threadIdx.x;
  int c = tid >> 12, p = tid & 4095, y = p >> 6, x = p & 63;
  const float* G = grid + (size_t)c * HW;
  float v[3][3];
#pragma unroll
  for (int dy = 0; dy < 3; ++dy)
#pragma unroll
    for (int dx = 0; dx < 3; ++dx) {
      int yy = y + dy - 1, xx = x + dx - 1;
      v[dy][dx] = ((unsigned)yy < 64u && (unsigned)xx < 64u) ? G[yy * WIM + xx] : 0.f;
    }
  float gx = -v[0][0] + v[0][2] - 2.f * v[1][0] + 2.f * v[1][2] - v[2][0] + v[2][2];
  float gy = -v[0][0] - 2.f * v[0][1] - v[0][2] + v[2][0] + 2.f * v[2][1] + v[2][2];
  float gmv = sqrtf(gx * gx + gy * gy + 1e-8f);
  float lv  = v[0][1] + v[1][0] + v[1][2] + v[2][1] - 4.f * v[1][1];
  gm[tid] = gmv;
  lap[tid] = lv;
  condX[(size_t)p * 512 + c]       = (_Float16)gmv;
  condX[(size_t)p * 512 + 256 + c] = (_Float16)v[1][1];
}

__global__ void k_update(const float* __restrict__ grid, const float* __restrict__ lap,
                         const float* __restrict__ cond, const float* __restrict__ mask,
                         const float* __restrict__ drp, float* __restrict__ x,
                         _Float16* __restrict__ xt) {
  int tid = blockIdx.x * 256 + threadIdx.x;
  int c = tid >> 12, p = tid & 4095;
  float dr = 1.f / (1.f + __expf(-drp[0]));
  float gv = grid[tid];
  float xv = (mask[p] > 0.5f) ? gv : gv + dr * cond[tid] * lap[tid];
  x[tid] = xv;
  xt[(size_t)p * 256 + c] = (_Float16)xv;
}

__global__ void k_gnstats(const float* __restrict__ r, float* __restrict__ st) {
  __shared__ float ss[256], sq[256];
  int g = blockIdx.x, t = threadIdx.x;
  const float* base = r + (size_t)g * 131072;
  float a = 0.f, b = 0.f;
  for (int i = t; i < 131072; i += 256) { float v = base[i]; a += v; b += v * v; }
  ss[t] = a; sq[t] = b;
  __syncthreads();
  for (int s2 = 128; s2 > 0; s2 >>= 1) {
    if (t < s2) { ss[t] += ss[t + s2]; sq[t] += sq[t + s2]; }
    __syncthreads();
  }
  if (t == 0) {
    float m = ss[0] * (1.f / 131072.f);
    st[2 * g] = m;
    st[2 * g + 1] = sq[0] * (1.f / 131072.f) - m * m;
  }
}

__global__ void k_gnapply(const float* __restrict__ r, const float* __restrict__ st,
                          const float* __restrict__ gma, const float* __restrict__ bta,
                          _Float16* __restrict__ out) {
  int tid = blockIdx.x * 256 + threadIdx.x;
  int c = tid >> 12, p = tid & 4095, g = c >> 5;
  float tv = (r[tid] - st[2 * g]) * rsqrtf(st[2 * g + 1] + 1e-5f) * gma[c] + bta[c];
  float ge = 0.5f * tv * (1.f + erff(tv * 0.70710678f));
  out[(size_t)p * 256 + c] = (_Float16)ge;
}

__global__ void k_residual(const float* __restrict__ x, const float* __restrict__ r2,
                           const float* __restrict__ orig, const float* __restrict__ mask,
                           const float* __restrict__ reswp, int s, float* __restrict__ grid) {
  int tid = blockIdx.x * 256 + threadIdx.x;
  int p = tid & 4095;
  float rw = 1.f / (1.f + __expf(-reswp[s]));
  float v = x[tid] + rw * r2[tid];
  grid[tid] = (mask[p] > 0.5f) ? orig[tid] : v;
}

__global__ void k_transpose16(const float* __restrict__ grid, _Float16* __restrict__ seqt) {
  int tid = blockIdx.x * 256 + threadIdx.x;
  int c = tid >> 12, p = tid & 4095;
  seqt[(size_t)p * 256 + c] = (_Float16)grid[tid];
}

// vT[(h*64+d)*4096 + s] = qkv[s*768 + 512 + h*64 + d]
__global__ void k_vt(const _Float16* __restrict__ qkv, _Float16* __restrict__ vT) {
  int tid = blockIdx.x * 256 + threadIdx.x;    // 2M threads
  int s = tid & 4095;
  int d = (tid >> 12) & 63;
  int h = tid >> 18;
  vT[tid] = qkv[(size_t)s * 768 + 512 + h * 64 + d];
}

__global__ void k_ln(const float* __restrict__ grid, const float* __restrict__ o2,
                     const float* __restrict__ g, const float* __restrict__ b,
                     _Float16* __restrict__ lnt) {
  int wave = threadIdx.x >> 5, lane = threadIdx.x & 31;
  int p = blockIdx.x * 8 + wave;
  float vals[8];
  float sum = 0.f;
#pragma unroll
  for (int j = 0; j < 8; ++j) {
    int c = lane * 8 + j;
    float v = grid[(size_t)c * HW + p] + o2[(size_t)p * 256 + c];
    vals[j] = v;
    sum += v;
  }
  for (int m = 16; m; m >>= 1) sum += __shfl_xor(sum, m);
  float mean = sum * (1.f / 256.f);
  float vs = 0.f;
#pragma unroll
  for (int j = 0; j < 8; ++j) { float d = vals[j] - mean; vs += d * d; }
  for (int m = 16; m; m >>= 1) vs += __shfl_xor(vs, m);
  float inv = rsqrtf(vs * (1.f / 256.f) + 1e-5f);
#pragma unroll
  for (int j = 0; j < 8; ++j) {
    int c = lane * 8 + j;
    lnt[(size_t)p * 256 + c] = (_Float16)((vals[j] - mean) * inv * g[c] + b[c]);
  }
}

// ---------------------------------------------------------------------------
// Generic WMMA GEMM: C[MxN] = act(A[MxK] * B[NxK]^T + bias)
// 8 waves/block, wave tile 16(M) x 64(N); B chunks double-buffered in LDS via
// async global->LDS DMA. Main loop stages unconditionally; last iter peeled.
// ---------------------------------------------------------------------------
__global__ __launch_bounds__(256) void k_gemm(
    const _Float16* __restrict__ A, int lda, const _Float16* __restrict__ B, int ldb,
    const float* __restrict__ bias, float* __restrict__ Cf, _Float16* __restrict__ Ch,
    int M, int N, int K, int act, int cm) {
  __shared__ _Float16 Bs[2][64 * 32];
  int t = threadIdx.x, wave = t >> 5, lane = t & 31;
  int mb = blockIdx.x * 128 + wave * 16;
  int nb = blockIdx.y * 64;
  int soff = (t >> 2) * 32 + ((t & 3) << 3);               // this thread's 16B slot
  const _Float16* sbase = B + (size_t)(nb + (t >> 2)) * ldb + ((t & 3) << 3);
  v8f a0 = {}, a1 = {}, a2 = {}, a3 = {};

  stage16B(sbase, &Bs[0][soff]);                           // prologue: chunk 0
  wait_stage();
  __syncthreads();
  int buf = 0;
  int kb = 0;
  for (; kb < K - 32; kb += 32) {                          // always stages next chunk
    stage16B(sbase + kb + 32, &Bs[buf ^ 1][soff]);
    v16h af = ldA(A, lda, mb, kb);
    a0 = do_wmma(af, ldBs(Bs[buf], 0), a0);
    a1 = do_wmma(af, ldBs(Bs[buf], 1), a1);
    a2 = do_wmma(af, ldBs(Bs[buf], 2), a2);
    a3 = do_wmma(af, ldBs(Bs[buf], 3), a3);
    wait_stage();
    __syncthreads();
    buf ^= 1;
  }
  {                                                        // peeled last chunk
    v16h af = ldA(A, lda, mb, kb);
    a0 = do_wmma(af, ldBs(Bs[buf], 0), a0);
    a1 = do_wmma(af, ldBs(Bs[buf], 1), a1);
    a2 = do_wmma(af, ldBs(Bs[buf], 2), a2);
    a3 = do_wmma(af, ldBs(Bs[buf], 3), a3);
  }

  int half8 = (lane >> 4) << 3;
  v8f accs[4] = {a0, a1, a2, a3};
#pragma unroll
  for (int nt = 0; nt < 4; ++nt) {
    int n = nb + (nt << 4) + (lane & 15);
    float bv = bias ? bias[n] : 0.f;
#pragma unroll
    for (int r = 0; r < 8; ++r) {
      int m = mb + r + half8;
      float v = accs[nt][r] + bv;
      if (act == 1) v = fmaxf(v, 0.f);
      else if (act == 2) v = 1.f / (1.f + __expf(-v));
      if (Ch) Ch[(size_t)m * N + n] = (_Float16)v;
      if (Cf) Cf[cm ? ((size_t)n * M + m) : ((size_t)m * N + n)] = v;
    }
  }
}

// A-fragment for the implicit-GEMM conv: chunk idx = tap*8 + kchunk.
DEVI v16h convA(const _Float16* Xt, int idx, int y, int x) {
  int lane = threadIdx.x & 31;
  int tap = idx >> 3, kb = (idx & 7) << 5;
  int ty = tap / 3, tx = tap % 3;
  int yy = y + ty - 1, xx = x + tx - 1;
  if (((unsigned)yy < 64u) && ((unsigned)xx < 64u)) {
    const _Float16* Arow = Xt + (size_t)(yy * WIM + xx) * 256 + kb + ((lane >> 4) << 3);
    HU u;
    u.q[0] = *(const v4u*)Arow;
    u.q[1] = *(const v4u*)(Arow + 16);
    return u.h;
  }
  return zfrag();
}

// ---------------------------------------------------------------------------
// Implicit-GEMM 3x3 conv: Out[c][p] = sum_tap sum_k Xt[nbr(p,tap)][k]*W[tap][c][k] + b
// 72 K-chunks (9 taps x 8), double-buffered async LDS staging, last iter peeled.
// ---------------------------------------------------------------------------
__global__ __launch_bounds__(256) void k_conv3(
    const _Float16* __restrict__ Xt, const _Float16* __restrict__ W,
    const float* __restrict__ bias, float* __restrict__ Out) {
  __shared__ _Float16 Bs[2][64 * 32];
  int t = threadIdx.x, wave = t >> 5, lane = t & 31;
  int mb = blockIdx.x * 128 + wave * 16;
  int nb = blockIdx.y * 64;
  int soff = (t >> 2) * 32 + ((t & 3) << 3);
  const _Float16* wbase = W + (size_t)(nb + (t >> 2)) * 256 + ((t & 3) << 3);
  v8f a0 = {}, a1 = {}, a2 = {}, a3 = {};
  int p = mb + (lane & 15);
  int y = p >> 6, x = p & 63;

  stage16B(wbase, &Bs[0][soff]);                           // chunk idx 0
  wait_stage();
  __syncthreads();
  int buf = 0;
  for (int idx = 0; idx < 71; ++idx) {                     // always stages idx+1
    int tap2 = (idx + 1) >> 3, kb2 = ((idx + 1) & 7) << 5;
    stage16B(wbase + (size_t)tap2 * 65536 + kb2, &Bs[buf ^ 1][soff]);
    v16h a = convA(Xt, idx, y, x);
    a0 = do_wmma(a, ldBs(Bs[buf], 0), a0);
    a1 = do_wmma(a, ldBs(Bs[buf], 1), a1);
    a2 = do_wmma(a, ldBs(Bs[buf], 2), a2);
    a3 = do_wmma(a, ldBs(Bs[buf], 3), a3);
    wait_stage();
    __syncthreads();
    buf ^= 1;
  }
  {                                                        // peeled chunk 71
    v16h a = convA(Xt, 71, y, x);
    a0 = do_wmma(a, ldBs(Bs[buf], 0), a0);
    a1 = do_wmma(a, ldBs(Bs[buf], 1), a1);
    a2 = do_wmma(a, ldBs(Bs[buf], 2), a2);
    a3 = do_wmma(a, ldBs(Bs[buf], 3), a3);
  }

  int half8 = (lane >> 4) << 3;
  v8f accs[4] = {a0, a1, a2, a3};
#pragma unroll
  for (int nt = 0; nt < 4; ++nt) {
    int n = nb + (nt << 4) + (lane & 15);
    float bv = bias[n];
#pragma unroll
    for (int r = 0; r < 8; ++r) {
      int pp = mb + r + half8;
      Out[(size_t)n * HW + pp] = accs[nt][r] + bv;
    }
  }
}

// ---------------------------------------------------------------------------
// Flash attention: per wave one 16-row Q tile for one head (S=4096, d=64).
// Scores via WMMA (K=64 -> 2 wmma per 16 keys), online softmax with 16-lane
// shuffles, P routed through LDS to re-layout as A-fragment, PV via WMMA.
// ---------------------------------------------------------------------------
__global__ __launch_bounds__(256) void k_flash(const _Float16* __restrict__ qkv,
                                               const _Float16* __restrict__ vT,
                                               _Float16* __restrict__ ot) {
  __shared__ _Float16 Pl[8][16 * 32];
  int wave = threadIdx.x >> 5, lane = threadIdx.x & 31;
  int head = blockIdx.y;
  int qbase = (blockIdx.x * 8 + wave) * 16;
  const _Float16* Qp = qkv + head * 64;
  const _Float16* Kp = qkv + 256 + head * 64;
  const _Float16* Vp = vT + (size_t)head * 64 * HW;
  _Float16* P = Pl[wave];

  v16h qa0 = ldA(Qp, 768, qbase, 0);
  v16h qa1 = ldA(Qp, 768, qbase, 32);
  v8f o0 = {}, o1 = {}, o2 = {}, o3 = {};
  float mst[8], lst[8];
#pragma unroll
  for (int r = 0; r < 8; ++r) { mst[r] = -1e30f; lst[r] = 0.f; }
  int half8 = (lane >> 4) << 3;

  for (int jb = 0; jb < HW; jb += 32) {
    v8f s0 = {}, s1 = {};
    s0 = do_wmma(qa0, ldB(Kp, 768, jb, 0), s0);
    s0 = do_wmma(qa1, ldB(Kp, 768, jb, 32), s0);
    s1 = do_wmma(qa0, ldB(Kp, 768, jb + 16, 0), s1);
    s1 = do_wmma(qa1, ldB(Kp, 768, jb + 16, 32), s1);
#pragma unroll
    for (int r = 0; r < 8; ++r) {
      float a = s0[r] * 0.125f, b = s1[r] * 0.125f;
      float rm = fmaxf(a, b);
      rm = fmaxf(rm, __shfl_xor(rm, 1));
      rm = fmaxf(rm, __shfl_xor(rm, 2));
      rm = fmaxf(rm, __shfl_xor(rm, 4));
      rm = fmaxf(rm, __shfl_xor(rm, 8));
      float mnew = fmaxf(mst[r], rm);
      float corr = __expf(mst[r] - mnew);
      float p0 = __expf(a - mnew), p1 = __expf(b - mnew);
      float rs = p0 + p1;
      rs += __shfl_xor(rs, 1);
      rs += __shfl_xor(rs, 2);
      rs += __shfl_xor(rs, 4);
      rs += __shfl_xor(rs, 8);
      lst[r] = lst[r] * corr + rs;
      mst[r] = mnew;
      o0[r] *= corr; o1[r] *= corr; o2[r] *= corr; o3[r] *= corr;
      int m = r + half8, n = lane & 15;
      P[m * 32 + n]      = (_Float16)p0;
      P[m * 32 + 16 + n] = (_Float16)p1;
    }
    asm volatile("s_wait_dscnt 0" ::: "memory");
    v16h pa;
    {
      const _Float16* pp = P + (lane & 15) * 32 + ((lane >> 4) << 3);
      HU u; u.q[0] = *(const v4u*)pp; u.q[1] = *(const v4u*)(pp + 16);
      pa = u.h;
    }
    o0 = do_wmma(pa, ldB(Vp, HW, 0,  jb), o0);
    o1 = do_wmma(pa, ldB(Vp, HW, 16, jb), o1);
    o2 = do_wmma(pa, ldB(Vp, HW, 32, jb), o2);
    o3 = do_wmma(pa, ldB(Vp, HW, 48, jb), o3);
  }
#pragma unroll
  for (int r = 0; r < 8; ++r) {
    int srow = qbase + r + half8;
    float inv = 1.f / lst[r];
    size_t base = (size_t)srow * 256 + head * 64 + (lane & 15);
    ot[base]      = (_Float16)(o0[r] * inv);
    ot[base + 16] = (_Float16)(o1[r] * inv);
    ot[base + 32] = (_Float16)(o2[r] * inv);
    ot[base + 48] = (_Float16)(o3[r] * inv);
  }
}

// ---------------------------------------------------------------------------
// Host driver
// ---------------------------------------------------------------------------
static inline unsigned nb256(long n) { return (unsigned)((n + 255) / 256); }

extern "C" void kernel_launch(void* const* d_in, const int* in_sizes, int n_in,
                              void* d_out, int out_size, void* d_ws, size_t ws_size,
                              hipStream_t stream) {
  (void)in_sizes; (void)n_in; (void)out_size; (void)ws_size;
  const float* sf      = (const float*)d_in[0];
  const int*   pos     = (const int*)d_in[1];
  const float* cond_w1 = (const float*)d_in[4];
  const float* cond_b1 = (const float*)d_in[5];
  const float* cond_w2 = (const float*)d_in[6];
  const float* cond_b2 = (const float*)d_in[7];
  const float* drate   = (const float*)d_in[8];
  const float* nl_w1   = (const float*)d_in[9];
  const float* nl_b1   = (const float*)d_in[10];
  const float* nl_g1   = (const float*)d_in[11];
  const float* nl_be1  = (const float*)d_in[12];
  const float* nl_w2   = (const float*)d_in[13];
  const float* nl_b2   = (const float*)d_in[14];
  const float* nl_resw = (const float*)d_in[15];
  const float* wqkv    = (const float*)d_in[16];
  const float* bqkv    = (const float*)d_in[17];
  const float* wo      = (const float*)d_in[18];
  const float* bo      = (const float*)d_in[19];
  const float* ln_g    = (const float*)d_in[20];
  const float* ln_b    = (const float*)d_in[21];
  const float* fp_w1   = (const float*)d_in[22];
  const float* fp_b1   = (const float*)d_in[23];
  const float* fp_g    = (const float*)d_in[24];
  const float* fp_be   = (const float*)d_in[25];
  const float* fp_w2   = (const float*)d_in[26];
  const float* fp_b2   = (const float*)d_in[27];

  char* W8 = (char*)d_ws;
  _Float16* wc1h  = (_Float16*)(W8 + OFF_WC1);
  _Float16* wc2h  = (_Float16*)(W8 + OFF_WC2);
  _Float16* wnl1h = (_Float16*)(W8 + OFF_WNL1);
  _Float16* wnl2h = (_Float16*)(W8 + OFF_WNL2);
  _Float16* wqkvh = (_Float16*)(W8 + OFF_WQKV);
  _Float16* wwoh  = (_Float16*)(W8 + OFF_WWO);
  _Float16* wfp1h = (_Float16*)(W8 + OFF_WFP1);
  _Float16* wfp2h = (_Float16*)(W8 + OFF_WFP2);
  float* grid = (float*)(W8 + OFF_GRID);
  float* orig = (float*)(W8 + OFF_ORIG);
  float* mask = (float*)(W8 + OFF_MASK);
  float* gst  = (float*)(W8 + OFF_GST);
  float* gmb  = (float*)(W8 + OFF_GM);
  float* lapb = (float*)(W8 + OFF_LAP);
  _Float16* condX = (_Float16*)(W8 + OFF_CX);
  float*    condf = (float*)(W8 + OFF_CX);      // alias (condX dead by then)
  _Float16* h16  = (_Float16*)(W8 + OFF_H16);
  float*    xb   = (float*)(W8 + OFF_X);
  _Float16* xt   = (_Float16*)(W8 + OFF_XT);
  float*    rb   = gmb;                          // alias
  float*    r2b  = lapb;                         // alias
  _Float16* rt   = (_Float16*)(W8 + OFF_RT);
  _Float16* seqt = (_Float16*)(W8 + OFF_SEQT);
  _Float16* qkvb = (_Float16*)(W8 + OFF_QKV);
  _Float16* vTb  = (_Float16*)(W8 + OFF_VT);
  _Float16* otb  = (_Float16*)(W8 + OFF_OT);
  float*    o2b  = (float*)(W8 + OFF_O2);
  _Float16* lntb = (_Float16*)(W8 + OFF_LNT);
  float*    yb   = (float*)(W8 + OFF_Y);
  _Float16* ytb  = (_Float16*)(W8 + OFF_YT);

  // --- weight prep (f32 -> f16, repacked) ---
  k_cast16<<<nb256(64 * 512), 256, 0, stream>>>(cond_w1, wc1h, 64 * 512);
  k_cast16<<<nb256(256 * 64), 256, 0, stream>>>(cond_w2, wc2h, 256 * 64);
  k_cast16<<<nb256(768 * 256), 256, 0, stream>>>(wqkv, wqkvh, 768 * 256);
  k_cast16<<<nb256(256 * 256), 256, 0, stream>>>(wo, wwoh, 256 * 256);
  k_cast16<<<nb256(256 * 256), 256, 0, stream>>>(fp_w2, wfp2h, 256 * 256);
  k_repack_conv<<<nb256(4 * 9 * 65536), 256, 0, stream>>>(nl_w1, wnl1h, 4 * 9 * 65536);
  k_repack_conv<<<nb256(4 * 9 * 65536), 256, 0, stream>>>(nl_w2, wnl2h, 4 * 9 * 65536);
  k_repack_conv<<<nb256(9 * 65536), 256, 0, stream>>>(fp_w1, wfp1h, 9 * 65536);

  // --- init grid / orig / mask (contiguous region) ---
  k_zero<<<nb256(2 * CF * HW + HW), 256, 0, stream>>>(grid, 2 * CF * HW + HW);
  k_scatter<<<nb256(2048 * 256), 256, 0, stream>>>(sf, pos, grid, orig, mask, 2048 * 256);

  // --- diffusion steps ---
  for (int s = 0; s < NSTEP; ++s) {
    k_stencil<<<nb256(CF * HW), 256, 0, stream>>>(grid, gmb, lapb, condX);
    // cond net: h = relu([gm;grid] @ W1^T + b1); cond = sigmoid(h @ W2^T + b2)
    k_gemm<<<dim3(32, 1), 256, 0, stream>>>(condX, 512, wc1h, 512, cond_b1,
                                            nullptr, h16, HW, HID, 512, 1, 0);
    k_gemm<<<dim3(32, 4), 256, 0, stream>>>(h16, 64, wc2h, 64, cond_b2,
                                            condf, nullptr, HW, CF, 64, 2, 1);
    k_update<<<nb256(CF * HW), 256, 0, stream>>>(grid, lapb, condf, mask, drate, xb, xt);
    // r1 = conv3x3(x, nl_w1[s]) + b1
    k_conv3<<<dim3(32, 4), 256, 0, stream>>>(xt, wnl1h + (size_t)s * 9 * 65536,
                                             nl_b1 + s * 256, rb);
    k_gnstats<<<8, 256, 0, stream>>>(rb, gst);
    k_gnapply<<<nb256(CF * HW), 256, 0, stream>>>(rb, gst, nl_g1 + s * 256,
                                                  nl_be1 + s * 256, rt);
    // r2 = conv3x3(gelu(gn(r1)), nl_w2[s]) + b2
    k_conv3<<<dim3(32, 4), 256, 0, stream>>>(rt, wnl2h + (size_t)s * 9 * 65536,
                                             nl_b2 + s * 256, r2b);
    k_residual<<<nb256(CF * HW), 256, 0, stream>>>(xb, r2b, orig, mask, nl_resw, s, grid);
  }

  // --- attention ---
  k_transpose16<<<nb256(CF * HW), 256, 0, stream>>>(grid, seqt);
  k_gemm<<<dim3(32, 12), 256, 0, stream>>>(seqt, 256, wqkvh, 256, bqkv,
                                           nullptr, qkvb, HW, 768, 256, 0, 0);
  k_vt<<<nb256(8 * 64 * HW), 256, 0, stream>>>(qkvb, vTb);
  k_flash<<<dim3(32, 8), 256, 0, stream>>>(qkvb, vTb, otb);
  k_gemm<<<dim3(32, 4), 256, 0, stream>>>(otb, 256, wwoh, 256, bo,
                                          o2b, nullptr, HW, CF, 256, 0, 0);
  k_ln<<<512, 256, 0, stream>>>(grid, o2b, ln_g, ln_b, lntb);

  // --- final projection ---
  k_conv3<<<dim3(32, 4), 256, 0, stream>>>(lntb, wfp1h, fp_b1, yb);
  k_gnstats<<<8, 256, 0, stream>>>(yb, gst);
  k_gnapply<<<nb256(CF * HW), 256, 0, stream>>>(yb, gst, fp_g, fp_be, ytb);
  k_gemm<<<dim3(32, 4), 256, 0, stream>>>(ytb, 256, wfp2h, 256, fp_b2,
                                          (float*)d_out, nullptr, HW, CF, 256, 0, 1);
}
